// SSIM_49211735278134
// MI455X (gfx1250) — matrix operand order repaired and checked by reading, not compiled, not run on hardware
//
#include <hip/hip_runtime.h>
#include <hip/hip_bf16.h>
#include <math.h>

// Problem constants (from reference)
#define BB 128
#define CC 5
#define SS 512
#define UU 256     // encoder hidden
#define RUU 512    // decoder hidden (2*U)
#define NE 1024    // encoder z width (4*U)
#define ND 2048    // decoder z width (4*RU)
#define NT_E 64    // encoder n-tiles (NE/16)
#define KT_E 8     // encoder k-tiles (UU/32)
#define NT_D 128   // decoder n-tiles (ND/16)
#define KT_D 16    // decoder k-tiles (RUU/32)

typedef __attribute__((ext_vector_type(16))) __bf16 v16bf;
typedef __attribute__((ext_vector_type(8)))  __bf16 v8bf;
typedef __attribute__((ext_vector_type(8)))  float  v8f;

// Global (address_space(1)) pointer flavors so anti-hoisted loads still lower
// to global_load_b128 (LOADcnt only) instead of flat_load_b128 (LOADcnt+DScnt).
typedef const __attribute__((address_space(1))) __bf16* gp_bf;
typedef const __attribute__((address_space(1))) v16bf*  gp_v16bf;

union ABu { v16bf v; v8bf h[2]; };

__device__ __forceinline__ float sigf(float x) { return 1.0f / (1.0f + __expf(-x)); }

// Break loop-invariance of a pointer: forces the compiler to keep dependent
// loads *inside* the surrounding loop (stream from L2) instead of hoisting
// them out and spilling the values to scratch. Re-asserts global addr space.
__device__ __forceinline__ gp_bf opaque_g(const __bf16* p) {
  unsigned long long a = (unsigned long long)p;
  asm volatile("" : "+s"(a));
  return (gp_bf)a;
}

// ---------------------------------------------------------------------------
// K1: extract xs[b][s][c] = x[b][c][s][0]; maskf[b][s] = (x[b][0][s][2] != 0)
// ---------------------------------------------------------------------------
__global__ void prep_kernel(const float* __restrict__ x,
                            float* __restrict__ xs, float* __restrict__ maskf) {
  int i = blockIdx.x * blockDim.x + threadIdx.x;  // over B*S
  if (i >= BB * SS) return;
  int b = i / SS, s = i % SS;
  float mv = x[((size_t)(b * CC + 0) * SS + s) * 3 + 2];
  maskf[i] = (mv != 0.0f) ? 1.0f : 0.0f;
  for (int c = 0; c < CC; ++c)
    xs[(size_t)i * CC + c] = x[((size_t)(b * CC + c) * SS + s) * 3 + 0];
}

// ---------------------------------------------------------------------------
// K2: pack a (K x N) f32 weight into bf16 WMMA B-fragment layout.
// Tile (kt, nt), lane l: n = nt*16 + (l&15); j=0..15 -> k = kt*32 + (l>>4)*16 + j
// (contiguous k order, not the ISA table order, matching the A loads; lanes 0-15 K=0..15, 16-31 K=16..31)
// ---------------------------------------------------------------------------
__global__ void pack_b_kernel(const float* __restrict__ W, __bf16* __restrict__ out,
                              int K, int N) {
  int gid = blockIdx.x * blockDim.x + threadIdx.x;
  int NT = N >> 4, KT = K >> 5;
  if (gid >= KT * NT * 32) return;
  int lane = gid & 31, tile = gid >> 5;
  int nt = tile % NT, kt = tile / NT;
  int n = nt * 16 + (lane & 15);
  int kbase = kt * 32 + ((lane >> 4) << 4);
  __bf16* dst = out + (size_t)gid * 16;
  for (int j = 0; j < 16; ++j)
    dst[j] = (__bf16)W[(size_t)(kbase + j) * N + n];
}

// ---------------------------------------------------------------------------
// K3: persistent bidirectional LSTM encoder. grid = (8 batch-tiles, 2 dirs).
// 256 threads = 8 waves; wave w owns u-tiles {2w, 2w+1}; per u-tile the 4
// gate n-tiles (i/f/g/o) accumulate in the same thread -> gate math in regs.
// kt-outer / gate-inner WMMA loop; B fragments streamed from L2 every step
// (opaque global pointer prevents hoist+scratch-spill, keeps LOADcnt-only).
// ---------------------------------------------------------------------------
__global__ __launch_bounds__(256) void encoder_kernel(
    const float* __restrict__ xs, const float* __restrict__ maskf,
    const float* __restrict__ Wf_k, const float* __restrict__ bf_,
    const float* __restrict__ Wb_k, const float* __restrict__ bb_,
    const __bf16* __restrict__ wfr_pk, const __bf16* __restrict__ wbr_pk,
    __bf16* __restrict__ enc) {
  const int dir = blockIdx.y;
  const int b0 = blockIdx.x * 16;
  const int tid = threadIdx.x;
  const int w = tid >> 5, lane = tid & 31, l15 = lane & 15, hi = lane >> 4;
  const float* Wk = dir ? Wb_k : Wf_k;
  const float* bias = dir ? bb_ : bf_;
  const __bf16* wpk = dir ? wbr_pk : wfr_pk;
  const int dOff = dir * UU;

  __shared__ __bf16 hbf[16][UU];   // 8 KB: h state, bf16, [m][k]
  __shared__ float xst[16][CC];
  __shared__ float mst[16];

  for (int i = tid; i < 16 * UU; i += 256) ((__bf16*)hbf)[i] = (__bf16)0.0f;

  float hreg[2][8], creg[2][8];
#pragma unroll
  for (int ut = 0; ut < 2; ++ut)
#pragma unroll
    for (int r = 0; r < 8; ++r) { hreg[ut][r] = 0.0f; creg[ut][r] = 0.0f; }

  // Hoist x->gate weights & bias for this thread's 8 n-positions (loop-invariant)
  float wkr[4][2][CC];
  float bres[4][2];
#pragma unroll
  for (int G = 0; G < 4; ++G)
#pragma unroll
    for (int ut = 0; ut < 2; ++ut) {
      int n_ = (G * 16 + (2 * w + ut)) * 16 + l15;
      bres[G][ut] = bias[n_];
#pragma unroll
      for (int c = 0; c < CC; ++c) wkr[G][ut][c] = Wk[(size_t)c * NE + n_];
    }
  __syncthreads();

  for (int st = 0; st < SS; ++st) {
    const int sIdx = dir ? (SS - 1 - st) : st;
    if (tid < 80) {
      int m = tid / CC, c = tid % CC;
      xst[m][c] = xs[((size_t)(b0 + m) * SS + sIdx) * CC + c];
    }
    if (tid < 16) mst[tid] = maskf[(size_t)(b0 + tid) * SS + sIdx];
    __syncthreads();  // xst/mst ready; prev-step hbf commits visible

    // z0 = bias + x_t @ Wk
    v8f acc[4][2];
#pragma unroll
    for (int G = 0; G < 4; ++G)
#pragma unroll
      for (int ut = 0; ut < 2; ++ut)
#pragma unroll
        for (int r = 0; r < 8; ++r) {
          int m = r + hi * 8;
          float z = bres[G][ut];
#pragma unroll
          for (int c = 0; c < CC; ++c) z += xst[m][c] * wkr[G][ut][c];
          acc[G][ut][r] = z;
        }

    // z += h @ Wr : kt outer (one A fragment live), 8 accumulators inner.
    // B fragments re-loaded from L2 each step (anti-hoist, global AS).
    gp_bf wpkL = opaque_g(wpk);
#pragma unroll
    for (int kt = 0; kt < KT_E; ++kt) {
      ABu va;
      va.h[0] = *(const v8bf*)&hbf[l15][kt * 32 + hi * 16];
      va.h[1] = *(const v8bf*)&hbf[l15][kt * 32 + hi * 16 + 8];
#pragma unroll
      for (int G = 0; G < 4; ++G)
#pragma unroll
        for (int ut = 0; ut < 2; ++ut) {
          const int ntile = G * 16 + (2 * w + ut);
          v16bf vb = *(gp_v16bf)(wpkL + (size_t)((kt * NT_E + ntile) * 32 + lane) * 16);
          acc[G][ut] = __builtin_amdgcn_wmma_f32_16x16x32_bf16(
              false, va.v, false, vb, (short)0, acc[G][ut], false, false);
        }
    }

    // Gate math + masked state update (registers only)
#pragma unroll
    for (int ut = 0; ut < 2; ++ut)
#pragma unroll
      for (int r = 0; r < 8; ++r) {
        int m = r + hi * 8;
        float zi = acc[0][ut][r], zf = acc[1][ut][r];
        float zg = acc[2][ut][r], zo = acc[3][ut][r];
        float cold = creg[ut][r], hold = hreg[ut][r];
        float cn = sigf(zf) * cold + sigf(zi) * tanhf(zg);
        float hn = sigf(zo) * tanhf(cn);
        float mk = mst[m];
        creg[ut][r] = cold + mk * (cn - cold);
        hreg[ut][r] = hold + mk * (hn - hold);
      }
    __syncthreads();  // all waves done reading hbf -> safe to commit

#pragma unroll
    for (int ut = 0; ut < 2; ++ut)
#pragma unroll
      for (int r = 0; r < 8; ++r) {
        int m = r + hi * 8;
        int u = (2 * w + ut) * 16 + l15;
        __bf16 hb = (__bf16)hreg[ut][r];
        hbf[m][u] = hb;
        enc[((size_t)(b0 + m) * SS + sIdx) * RUU + dOff + u] = hb;
      }
    __syncthreads();
  }
}

// ---------------------------------------------------------------------------
// K4: enc_scores[b][s] = b_attn + sum_d enc[b][s][d] * We[d]
// One wave per (b,s) row: coalesced lane-strided loads + shuffle reduction.
// ---------------------------------------------------------------------------
__global__ __launch_bounds__(256) void scores_kernel(const __bf16* __restrict__ enc,
                                                     const float* __restrict__ W_attn,
                                                     const float* __restrict__ b_attn,
                                                     float* __restrict__ scores) {
  int w = threadIdx.x >> 5, lane = threadIdx.x & 31;
  int row = blockIdx.x * 8 + w;  // b*S + s
  if (row >= BB * SS) return;
  const __bf16* e = enc + (size_t)row * RUU;
  float acc = 0.0f;
#pragma unroll
  for (int it = 0; it < RUU / 32; ++it) {
    int d = lane + it * 32;
    acc += (float)e[d] * W_attn[CC + d];
  }
#pragma unroll
  for (int o = 16; o > 0; o >>= 1) acc += __shfl_xor(acc, o, 32);
  if (lane == 0) scores[row] = acc + b_attn[0];
}

// ---------------------------------------------------------------------------
// K5: persistent attention decoder. grid = 8 batch-tiles x 256 threads.
// Per step: softmax stats (wave-reduce) -> attention einsum (enc in L2) ->
// WMMA z = x@Wd_k + h@Wd_r + bd (wave w owns u-tiles 4w..4w+3, gate-aligned,
// kt-outer WMMA loop, B fragments streamed from L2 per n-tile group) ->
// gates with c=attn -> half-wave-parallel dense for output + next x_t.
// ---------------------------------------------------------------------------
__global__ __launch_bounds__(256) void decoder_kernel(
    const float* __restrict__ xs, const float* __restrict__ scores,
    const __bf16* __restrict__ enc, const __bf16* __restrict__ wdr_pk,
    const float* __restrict__ Wd_k, const float* __restrict__ bd,
    const float* __restrict__ W_attn, const float* __restrict__ W_dense,
    const float* __restrict__ b_dense, float* __restrict__ out) {
  const int b0 = blockIdx.x * 16;
  const int tid = threadIdx.x;
  const int w = tid >> 5, lane = tid & 31, l15 = lane & 15, hi = lane >> 4;

  __shared__ __bf16 hid[16][RUU];    // 16 KB: hidden (bf16, [m][k])
  __shared__ float attnS[16][RUU];   // 32 KB
  __shared__ float xtS[16][CC];
  __shared__ float scS[16];
  __shared__ float mxS[16];
  __shared__ float zinvS[16];

  for (int i = tid; i < 16 * RUU; i += 256) {
    int m = i >> 9, k = i & 511;
    hid[m][k] = enc[((size_t)(b0 + m) * SS + (SS - 1)) * RUU + k];  // hidden0
  }
  if (tid < 80) {
    int m = tid / CC, c = tid % CC;
    xtS[m][c] = xs[((size_t)(b0 + m) * SS + 0) * CC + c];  // x0
  }
  __syncthreads();

  for (int t = 0; t < SS - 1; ++t) {
    // (a) per-row score bias: x_t @ Wx
    if (tid < 16) {
      float s0 = 0.0f;
#pragma unroll
      for (int c = 0; c < CC; ++c) s0 += xtS[tid][c] * W_attn[c];
      scS[tid] = s0;
    }
    __syncthreads();

    // (b) softmax stats: wave w handles rows 2w, 2w+1
    for (int rr = 0; rr < 2; ++rr) {
      int m = 2 * w + rr;
      const float* srow = scores + (size_t)(b0 + m) * SS;
      float sc = scS[m];
      float mx = -1e30f;
      for (int it = 0; it < 16; ++it) mx = fmaxf(mx, srow[lane + it * 32] + sc);
      for (int o = 16; o > 0; o >>= 1) mx = fmaxf(mx, __shfl_xor(mx, o, 32));
      float sum = 0.0f;
      for (int it = 0; it < 16; ++it) sum += __expf(srow[lane + it * 32] + sc - mx);
      for (int o = 16; o > 0; o >>= 1) sum += __shfl_xor(sum, o, 32);
      if (lane == 0) { mxS[m] = mx; zinvS[m] = 1.0f / sum; }
    }
    __syncthreads();

    // (c) attention: thread (m = tid/16, 32-wide d chunk); enc is L2-resident
    {
      int m = tid >> 4, d0 = (tid & 15) * 32;
      float a[32];
#pragma unroll
      for (int j = 0; j < 32; ++j) a[j] = 0.0f;
      const float* srow = scores + (size_t)(b0 + m) * SS;
      float sc = scS[m], mx = mxS[m];
      const __bf16* erow = enc + (size_t)(b0 + m) * SS * RUU + d0;
      for (int s = 0; s < SS; ++s) {
        float e = __expf(srow[s] + sc - mx);
        const __bf16* er = erow + (size_t)s * RUU;
#pragma unroll
        for (int j = 0; j < 32; ++j) a[j] += e * (float)er[j];
      }
      float zi = zinvS[m];
#pragma unroll
      for (int j = 0; j < 32; ++j) attnS[m][d0 + j] = a[j] * zi;
    }
    __syncthreads();

    // (e) WMMA gates; wave w owns u-tiles 4w..4w+3; h_new deferred in regs
    float hnew[4][8];
    for (int q = 0; q < 4; ++q) {
      const int tu = 4 * w + q;
      v8f acc[4];
#pragma unroll
      for (int G = 0; G < 4; ++G) {
        int n = G * RUU + tu * 16 + l15;
        float wk[CC];
#pragma unroll
        for (int c = 0; c < CC; ++c) wk[c] = Wd_k[(size_t)c * ND + n];
        float bz = bd[n];
#pragma unroll
        for (int r = 0; r < 8; ++r) {
          int m = r + hi * 8;
          float z = bz;
#pragma unroll
          for (int c = 0; c < CC; ++c) z += xtS[m][c] * wk[c];
          acc[G][r] = z;
        }
      }
      // B fragments re-loaded from L2 per n-tile group (anti-hoist, global AS)
      gp_bf wdrL = opaque_g(wdr_pk);
#pragma unroll
      for (int kt = 0; kt < KT_D; ++kt) {
        ABu va;
        va.h[0] = *(const v8bf*)&hid[l15][kt * 32 + hi * 16];
        va.h[1] = *(const v8bf*)&hid[l15][kt * 32 + hi * 16 + 8];
#pragma unroll
        for (int G = 0; G < 4; ++G) {
          const int ntile = G * 32 + tu;
          v16bf vb = *(gp_v16bf)(wdrL + (size_t)((kt * NT_D + ntile) * 32 + lane) * 16);
          acc[G] = __builtin_amdgcn_wmma_f32_16x16x32_bf16(
              false, va.v, false, vb, (short)0, acc[G], false, false);
        }
      }
#pragma unroll
      for (int r = 0; r < 8; ++r) {
        int m = r + hi * 8;
        int u = tu * 16 + l15;
        float zi_ = acc[0][r], zf_ = acc[1][r], zg_ = acc[2][r], zo_ = acc[3][r];
        float cn = sigf(zf_) * attnS[m][u] + sigf(zi_) * tanhf(zg_);  // c = attn
        hnew[q][r] = sigf(zo_) * tanhf(cn);
      }
    }
    __syncthreads();  // all waves done reading hid -> safe to commit h_new
#pragma unroll
    for (int q = 0; q < 4; ++q)
#pragma unroll
      for (int r = 0; r < 8; ++r) {
        int m = r + hi * 8;
        int u = (4 * w + q) * 16 + l15;
        hid[m][u] = (__bf16)hnew[q][r];
      }
    __syncthreads();

    // (f) dense: x_new = [h_new, attn] @ W_dense + b_dense
    // 16 half-wave groups (one per row m), 16-lane shuffle reduction.
    {
      int m = tid >> 4, l16 = tid & 15;
      float part[CC];
#pragma unroll
      for (int c = 0; c < CC; ++c) part[c] = 0.0f;
      for (int it = 0; it < RUU / 16; ++it) {
        int u = l16 + it * 16;
        float hv = (float)hid[m][u];
        float av = attnS[m][u];
        const float* wd0 = W_dense + (size_t)u * CC;
        const float* wd1 = W_dense + (size_t)(RUU + u) * CC;
#pragma unroll
        for (int c = 0; c < CC; ++c) part[c] += hv * wd0[c] + av * wd1[c];
      }
#pragma unroll
      for (int c = 0; c < CC; ++c)
#pragma unroll
        for (int o = 8; o > 0; o >>= 1) part[c] += __shfl_xor(part[c], o, 32);
      if (l16 < CC) {
        float accd = b_dense[l16];
#pragma unroll
        for (int c = 0; c < CC; ++c)
          if (l16 == c) accd += part[c];
        out[((size_t)(b0 + m) * CC + l16) * (SS - 1) + t] = accd;
        xtS[m][l16] = accd;
      }
    }
    __syncthreads();
  }
}

// ---------------------------------------------------------------------------
extern "C" void kernel_launch(void* const* d_in, const int* in_sizes, int n_in,
                              void* d_out, int out_size, void* d_ws, size_t ws_size,
                              hipStream_t stream) {
  const float* x       = (const float*)d_in[0];
  const float* Wf_k    = (const float*)d_in[1];
  const float* Wf_r    = (const float*)d_in[2];
  const float* bf_     = (const float*)d_in[3];
  const float* Wb_k    = (const float*)d_in[4];
  const float* Wb_r    = (const float*)d_in[5];
  const float* bb_     = (const float*)d_in[6];
  const float* Wd_k    = (const float*)d_in[7];
  const float* Wd_r    = (const float*)d_in[8];
  const float* bd      = (const float*)d_in[9];
  const float* W_attn  = (const float*)d_in[10];
  const float* b_attn  = (const float*)d_in[11];
  const float* W_dense = (const float*)d_in[12];
  const float* b_dense = (const float*)d_in[13];
  float* out = (float*)d_out;
  (void)in_sizes; (void)n_in; (void)out_size; (void)ws_size;

  char* ws = (char*)d_ws;
  size_t off = 0;
  auto carve = [&](size_t bytes) { void* p = ws + off; off += (bytes + 255) & ~(size_t)255; return p; };
  float*  xs     = (float*) carve((size_t)BB * SS * CC * 4);   // 1.3 MB
  float*  maskf  = (float*) carve((size_t)BB * SS * 4);        // 256 KB
  __bf16* wfrp   = (__bf16*)carve((size_t)UU * NE * 2);        // 512 KB
  __bf16* wbrp   = (__bf16*)carve((size_t)UU * NE * 2);        // 512 KB
  __bf16* wdrp   = (__bf16*)carve((size_t)RUU * ND * 2);       // 2 MB
  __bf16* enc    = (__bf16*)carve((size_t)BB * SS * RUU * 2);  // 64 MB (L2-resident)
  float*  scores = (float*) carve((size_t)BB * SS * 4);        // 256 KB

  prep_kernel<<<(BB * SS + 255) / 256, 256, 0, stream>>>(x, xs, maskf);
  pack_b_kernel<<<(KT_E * NT_E * 32 + 255) / 256, 256, 0, stream>>>(Wf_r, wfrp, UU, NE);
  pack_b_kernel<<<(KT_E * NT_E * 32 + 255) / 256, 256, 0, stream>>>(Wb_r, wbrp, UU, NE);
  pack_b_kernel<<<(KT_D * NT_D * 32 + 255) / 256, 256, 0, stream>>>(Wd_r, wdrp, RUU, ND);
  encoder_kernel<<<dim3(8, 2), 256, 0, stream>>>(xs, maskf, Wf_k, bf_, Wb_k, bb_,
                                                 wfrp, wbrp, enc);
  scores_kernel<<<(BB * SS + 7) / 8, 256, 0, stream>>>(enc, W_attn, b_attn, scores);
  decoder_kernel<<<8, 256, 0, stream>>>(xs, scores, enc, wdrp, Wd_k, bd,
                                        W_attn, W_dense, b_dense, out);
}
